// MultiHeadAttention_84061099917631
// MI455X (gfx1250) — matrix operand order repaired and checked by reading, not compiled, not run on hardware
//
#include <hip/hip_runtime.h>
#include <hip/hip_bf16.h>

// MHA forward: B=4, S=2048, D=1024, H=16, DH=64 (fp32 in/out).
// Strategy: all GEMMs on v_wmma_f32_16x16x32_bf16 (bf16 operands, f32 accum),
// flash-attention streaming softmax so the (B,H,S,S) score tensor never
// touches HBM. K tiles DMA'd into LDS with global_load_async_to_lds_b128
// (ASYNCcnt), V tiles transposed in flight, shared by 8 waves per block.

typedef __attribute__((ext_vector_type(16))) __bf16       bf16x16;
typedef __attribute__((ext_vector_type(8)))  float        f32x8;
typedef __attribute__((ext_vector_type(4)))  float        f32x4;
typedef __attribute__((ext_vector_type(4)))  unsigned int u32x4;
typedef __attribute__((ext_vector_type(8)))  unsigned int u32x8;

static __device__ __forceinline__ unsigned short f2bf(float f) {
  union { float f; unsigned u; } c; c.f = f;
  unsigned u = c.u + 0x7FFFu + ((c.u >> 16) & 1u);   // round-to-nearest-even
  return (unsigned short)(u >> 16);
}

static __device__ __forceinline__ bf16x16 frag_from(u32x4 lo, u32x4 hi) {
  u32x8 t;
  t[0] = lo[0]; t[1] = lo[1]; t[2] = lo[2]; t[3] = lo[3];
  t[4] = hi[0]; t[5] = hi[1]; t[6] = hi[2]; t[7] = hi[3];
  return __builtin_bit_cast(bf16x16, t);
}

static __device__ __forceinline__ f32x8 wmma_bf16(bf16x16 a, bf16x16 b, f32x8 c) {
  return __builtin_amdgcn_wmma_f32_16x16x32_bf16(false, a, false, b, (short)0, c,
                                                 false, false);
}

// ---------------------------------------------------------------------------
// Shared GEMM core: C(128x64) += A(128x1024) * B(1024x64)
// A: f32, lda=1024 (row-major).  B: f32, row stride ldb, col base already applied.
// 256 threads = 8 waves; wave w computes rows [w*16, w*16+16), 4 N-tiles of 16.
// X tile converted to bf16 in LDS [128][40]; B tile transposed bf16 [64][40].
// ---------------------------------------------------------------------------
static __device__ __forceinline__ void gemm_core(
    const float* __restrict__ Abase, const float* __restrict__ Bbase, int ldb,
    unsigned short* Xs, unsigned short* Ws, f32x8 acc[4]) {
  const int t     = threadIdx.x;
  const int lane  = t & 31;
  const int wave  = t >> 5;
  const int l16   = lane & 15;
  const int koffA = (lane < 16) ? 0 : 8;    // A frag: K {0-7,16-23} / {8-15,24-31}
  const int koffB = (lane < 16) ? 0 : 8;    // B frag: K {0-7,16-23} / {8-15,24-31}
  const int XSTR  = 40;

  f32x8 z = {0.f, 0.f, 0.f, 0.f, 0.f, 0.f, 0.f, 0.f};
#pragma unroll
  for (int nt = 0; nt < 4; nt++) acc[nt] = z;

  for (int kt = 0; kt < 1024; kt += 32) {
    // --- stage X tile (128x32 f32 -> bf16 LDS), 16 contiguous elems/thread ---
    {
      int idx = t * 16;
      int row = idx >> 5;          // 0..127 (2 threads/row)
      int col = idx & 31;          // 0 or 16
      const float* src = Abase + (size_t)row * 1024 + kt + col;
      f32x4 a0 = *(const f32x4*)(src + 0);
      f32x4 a1 = *(const f32x4*)(src + 4);
      f32x4 a2 = *(const f32x4*)(src + 8);
      f32x4 a3 = *(const f32x4*)(src + 12);
      u32x4 s0, s1;
      s0[0] = f2bf(a0[0]) | ((unsigned)f2bf(a0[1]) << 16);
      s0[1] = f2bf(a0[2]) | ((unsigned)f2bf(a0[3]) << 16);
      s0[2] = f2bf(a1[0]) | ((unsigned)f2bf(a1[1]) << 16);
      s0[3] = f2bf(a1[2]) | ((unsigned)f2bf(a1[3]) << 16);
      s1[0] = f2bf(a2[0]) | ((unsigned)f2bf(a2[1]) << 16);
      s1[1] = f2bf(a2[2]) | ((unsigned)f2bf(a2[3]) << 16);
      s1[2] = f2bf(a3[0]) | ((unsigned)f2bf(a3[1]) << 16);
      s1[3] = f2bf(a3[2]) | ((unsigned)f2bf(a3[3]) << 16);
      *(u32x4*)&Xs[row * XSTR + col]     = s0;
      *(u32x4*)&Xs[row * XSTR + col + 8] = s1;
    }
    // --- stage B tile (32x64 f32) transposed -> bf16 LDS [n][k] ---
    {
      int widx = t * 8;
      int wk   = widx >> 6;        // 0..31
      int wn   = widx & 63;        // multiple of 8
      const float* wsrc = Bbase + (size_t)(kt + wk) * ldb + wn;
      f32x4 b0 = *(const f32x4*)(wsrc + 0);
      f32x4 b1 = *(const f32x4*)(wsrc + 4);
#pragma unroll
      for (int i = 0; i < 4; i++) Ws[(wn + i) * XSTR + wk]     = f2bf(b0[i]);
#pragma unroll
      for (int i = 0; i < 4; i++) Ws[(wn + 4 + i) * XSTR + wk] = f2bf(b1[i]);
    }
    __syncthreads();

    // A fragment: 16x32, lane l16 = row
    const int arow = wave * 16 + l16;
    u32x4 alo = *(const u32x4*)&Xs[arow * XSTR + koffA];
    u32x4 ahi = *(const u32x4*)&Xs[arow * XSTR + koffA + 16];
    bf16x16 af = frag_from(alo, ahi);

#pragma unroll
    for (int nt = 0; nt < 4; nt++) {
      const int brow = nt * 16 + l16;     // column of B == row of transposed LDS
      u32x4 blo = *(const u32x4*)&Ws[brow * XSTR + koffB];
      u32x4 bhi = *(const u32x4*)&Ws[brow * XSTR + koffB + 16];
      acc[nt] = wmma_bf16(af, frag_from(blo, bhi), acc[nt]);
    }
    __syncthreads();
  }
}

// ---------------------------------------------------------------------------
// Kernel 1: fused Q/K/V projections -> bf16 (B,H,S,DH).  grid=(64,H,3)
// ---------------------------------------------------------------------------
__global__ __launch_bounds__(256) void qkv_gemm_kernel(
    const float* __restrict__ x,
    const float* __restrict__ Wq, const float* __restrict__ Wk,
    const float* __restrict__ Wv,
    unsigned short* __restrict__ Qo, unsigned short* __restrict__ Ko,
    unsigned short* __restrict__ Vo) {
  __shared__ unsigned short Xs[128 * 40];
  __shared__ unsigned short Ws[64 * 40];
  const int h   = blockIdx.y;
  const int sel = blockIdx.z;
  const float* W       = (sel == 0) ? Wq : (sel == 1 ? Wk : Wv);
  unsigned short* Out  = (sel == 0) ? Qo : (sel == 1 ? Ko : Vo);
  const int row0 = blockIdx.x * 128;

  f32x8 acc[4];
  gemm_core(x + (size_t)row0 * 1024, W + (size_t)h * 1024 * 64, 64, Xs, Ws, acc);

  const int t = threadIdx.x, lane = t & 31, wave = t >> 5;
  const int mh = (lane >> 4) * 8, l16 = lane & 15;
#pragma unroll
  for (int nt = 0; nt < 4; nt++)
#pragma unroll
    for (int r = 0; r < 8; r++) {
      int grow = row0 + wave * 16 + mh + r;
      int b = grow >> 11;            // / 2048
      int s = grow & 2047;
      size_t o = ((size_t)(b * 16 + h) * 2048 + s) * 64 + nt * 16 + l16;
      Out[o] = f2bf(acc[nt][r]);
    }
}

// ---------------------------------------------------------------------------
// Kernel 2: flash attention.  grid=(S/128, B*H); block = 256 (8 waves).
// Each wave owns 16 query rows; K tile [key][d] via async DMA to LDS,
// V tile transposed [d][key] via VGPR path.
// ---------------------------------------------------------------------------
__global__ __launch_bounds__(256) void flash_attn_kernel(
    const unsigned short* __restrict__ Q, const unsigned short* __restrict__ K,
    const unsigned short* __restrict__ V, float* __restrict__ Oout) {
  __shared__ unsigned short Ks[64 * 72];
  __shared__ unsigned short Vs[64 * 72];
  __shared__ unsigned short Ps[8 * 16 * 72];

  const int t = threadIdx.x, lane = t & 31, wave = t >> 5;
  const int l16 = lane & 15, mh = (lane >> 4) * 8;
  const int koffA = (lane < 16) ? 0 : 8;
  const int koffB = (lane < 16) ? 0 : 8;
  const int bh = blockIdx.y;
  const int q0 = blockIdx.x * 128;
  const size_t base = (size_t)bh * 2048 * 64;

  // Q fragments: 16x64 as two 16x32 A-frags, loaded straight from global bf16
  bf16x16 qf[2];
  {
    const unsigned short* qrow = Q + base + (size_t)(q0 + wave * 16 + l16) * 64;
#pragma unroll
    for (int kh = 0; kh < 2; kh++) {
      u32x4 lo = *(const u32x4*)(qrow + kh * 32 + koffA);
      u32x4 hi = *(const u32x4*)(qrow + kh * 32 + koffA + 16);
      qf[kh] = frag_from(lo, hi);
    }
  }

  f32x8 z = {0.f, 0.f, 0.f, 0.f, 0.f, 0.f, 0.f, 0.f};
  f32x8 o[4];
#pragma unroll
  for (int nt = 0; nt < 4; nt++) o[nt] = z;
  float mrow[8], lrow[8];
#pragma unroll
  for (int r = 0; r < 8; r++) { mrow[r] = -3.0e38f; lrow[r] = 0.f; }

  for (int k0 = 0; k0 < 2048; k0 += 64) {
    // --- cooperative K/V tile loads (64 keys x 64 d), 16 bf16 per thread ---
    {
      int idx = t * 16;
      int key = idx >> 6;            // 0..63
      int d0  = idx & 63;            // 0,16,32,48
      // K tile: async DMA global -> LDS, 2 x 16B per lane, no VGPR round-trip.
      // Generic pointer to __shared__ carries the LDS byte offset in its low
      // 32 bits (flat aperture); INST_OFFSET is added
      // to both the LDS and global addresses per the async pseudocode.
      unsigned kdst = (unsigned)(size_t)&Ks[key * 72 + d0];
      const unsigned short* kp = K + base + (size_t)(k0 + key) * 64 + d0;
      asm volatile("global_load_async_to_lds_b128 %0, %1, off"
                   :: "v"(kdst), "v"(kp) : "memory");
      asm volatile("global_load_async_to_lds_b128 %0, %1, off offset:16"
                   :: "v"(kdst), "v"(kp) : "memory");
      // V tile: transposed in flight (async raw copy cannot express this)
      const unsigned short* vp = V + base + (size_t)(k0 + key) * 64 + d0;
      union { u32x4 v; unsigned short s[8]; } c0, c1;
      c0.v = *(const u32x4*)(vp);
      c1.v = *(const u32x4*)(vp + 8);
#pragma unroll
      for (int i = 0; i < 8; i++) Vs[(d0 + i) * 72 + key]     = c0.s[i];
#pragma unroll
      for (int i = 0; i < 8; i++) Vs[(d0 + 8 + i) * 72 + key] = c1.s[i];
      if (k0 + 64 < 2048) {          // prefetch next tile into L2
        __builtin_prefetch(K + base + (size_t)(k0 + 64 + key) * 64 + d0, 0, 1);
        __builtin_prefetch(V + base + (size_t)(k0 + 64 + key) * 64 + d0, 0, 1);
      }
    }
    // Drain this wave's async DMAs before the workgroup barrier so every wave
    // sees a complete K tile after the barrier.
    asm volatile("s_wait_asynccnt 0x0" ::: "memory");
    __syncthreads();

    // --- scores S = Q K^T * 1/sqrt(64): 4 tiles of 16 keys ---
    f32x8 sc[4];
#pragma unroll
    for (int nt = 0; nt < 4; nt++) {
      f32x8 a = z;
#pragma unroll
      for (int kh = 0; kh < 2; kh++) {
        const unsigned short* bp = &Ks[(nt * 16 + l16) * 72 + kh * 32 + koffB];
        u32x4 lo = *(const u32x4*)(bp);
        u32x4 hi = *(const u32x4*)(bp + 16);
        a = wmma_bf16(qf[kh], frag_from(lo, hi), a);
      }
      sc[nt] = a * 0.125f;
    }

    // --- streaming softmax update (rows live in 16-lane halves) ---
    float tmax[8];
#pragma unroll
    for (int r = 0; r < 8; r++)
      tmax[r] = fmaxf(fmaxf(sc[0][r], sc[1][r]), fmaxf(sc[2][r], sc[3][r]));
#pragma unroll
    for (int off = 1; off < 16; off <<= 1)
#pragma unroll
      for (int r = 0; r < 8; r++)
        tmax[r] = fmaxf(tmax[r], __shfl_xor(tmax[r], off, 32));
    float corr[8], rsum[8];
#pragma unroll
    for (int r = 0; r < 8; r++) {
      float nm = fmaxf(mrow[r], tmax[r]);
      corr[r] = __expf(mrow[r] - nm);
      mrow[r] = nm;
      rsum[r] = 0.f;
    }
#pragma unroll
    for (int nt = 0; nt < 4; nt++)
#pragma unroll
      for (int r = 0; r < 8; r++) {
        float p = __expf(sc[nt][r] - mrow[r]);
        sc[nt][r] = p;
        rsum[r] += p;
      }
#pragma unroll
    for (int off = 1; off < 16; off <<= 1)
#pragma unroll
      for (int r = 0; r < 8; r++) rsum[r] += __shfl_xor(rsum[r], off, 32);
#pragma unroll
    for (int r = 0; r < 8; r++) lrow[r] = lrow[r] * corr[r] + rsum[r];
#pragma unroll
    for (int nt = 0; nt < 4; nt++)
#pragma unroll
      for (int r = 0; r < 8; r++) o[nt][r] *= corr[r];

    // --- transpose P through per-wave LDS scratch (C-layout -> A-layout) ---
    unsigned short* pw = &Ps[wave * 16 * 72];
#pragma unroll
    for (int nt = 0; nt < 4; nt++)
#pragma unroll
      for (int r = 0; r < 8; r++)
        pw[(mh + r) * 72 + nt * 16 + l16] = f2bf(sc[nt][r]);
    asm volatile("s_wait_dscnt 0" ::: "memory");   // same-wave LDS RAW; in-order pipe

    bf16x16 pf[2];
    const unsigned short* pr = &Ps[(wave * 16 + l16) * 72];
#pragma unroll
    for (int kh = 0; kh < 2; kh++) {
      u32x4 lo = *(const u32x4*)(pr + kh * 32 + koffA);
      u32x4 hi = *(const u32x4*)(pr + kh * 32 + koffA + 16);
      pf[kh] = frag_from(lo, hi);
    }

    // --- O += P V : B-frags from transposed V tile ---
#pragma unroll
    for (int nt = 0; nt < 4; nt++)
#pragma unroll
      for (int kh = 0; kh < 2; kh++) {
        const unsigned short* bp = &Vs[(nt * 16 + l16) * 72 + kh * 32 + koffB];
        u32x4 lo = *(const u32x4*)(bp);
        u32x4 hi = *(const u32x4*)(bp + 16);
        o[nt] = wmma_bf16(pf[kh], frag_from(lo, hi), o[nt]);
      }
    __syncthreads();
  }

  // --- finalize: O /= l, write f32 (B,S,H*DH) for output projection ---
  const int b = bh >> 4, h = bh & 15;
  float inv[8];
#pragma unroll
  for (int r = 0; r < 8; r++) inv[r] = 1.0f / lrow[r];
#pragma unroll
  for (int nt = 0; nt < 4; nt++)
#pragma unroll
    for (int r = 0; r < 8; r++) {
      int srow = q0 + wave * 16 + mh + r;
      size_t off = ((size_t)(b * 2048 + srow)) * 1024 + h * 64 + nt * 16 + l16;
      Oout[off] = o[nt][r] * inv[r];
    }
}

// ---------------------------------------------------------------------------
// Kernel 3: output projection (8192x1024) @ Wo(1024x1024) -> f32 d_out
// ---------------------------------------------------------------------------
__global__ __launch_bounds__(256) void out_gemm_kernel(
    const float* __restrict__ attn, const float* __restrict__ Wo,
    float* __restrict__ out) {
  __shared__ unsigned short Xs[128 * 40];
  __shared__ unsigned short Ws[64 * 40];
  const int row0 = blockIdx.x * 128;
  const int n0   = blockIdx.y * 64;

  f32x8 acc[4];
  gemm_core(attn + (size_t)row0 * 1024, Wo + n0, 1024, Xs, Ws, acc);

  const int t = threadIdx.x, lane = t & 31, wave = t >> 5;
  const int mh = (lane >> 4) * 8, l16 = lane & 15;
#pragma unroll
  for (int nt = 0; nt < 4; nt++)
#pragma unroll
    for (int r = 0; r < 8; r++) {
      size_t grow = (size_t)(row0 + wave * 16 + mh + r);
      out[grow * 1024 + n0 + nt * 16 + l16] = acc[nt][r];
    }
}

// ---------------------------------------------------------------------------
extern "C" void kernel_launch(void* const* d_in, const int* in_sizes, int n_in,
                              void* d_out, int out_size, void* d_ws,
                              size_t ws_size, hipStream_t stream) {
  const float* x  = (const float*)d_in[0];
  const float* Wq = (const float*)d_in[1];
  const float* Wk = (const float*)d_in[2];
  const float* Wv = (const float*)d_in[3];
  const float* Wo = (const float*)d_in[4];

  const size_t qkvElems = (size_t)4 * 16 * 2048 * 64;   // 8,388,608 per tensor
  unsigned short* Qws = (unsigned short*)d_ws;
  unsigned short* Kws = Qws + qkvElems;
  unsigned short* Vws = Kws + qkvElems;
  float* attn = (float*)(Vws + qkvElems);               // 48 MB offset, aligned
  float* out  = (float*)d_out;

  dim3 g1(64, 16, 3);   // (B*S/128, H, {q,k,v})
  qkv_gemm_kernel<<<g1, 256, 0, stream>>>(x, Wq, Wk, Wv, Qws, Kws, Vws);

  dim3 g2(16, 64);      // (S/128, B*H)
  flash_attn_kernel<<<g2, 256, 0, stream>>>(Qws, Kws, Vws, attn);

  dim3 g3(64, 16);      // (B*S/128, D/64)
  out_gemm_kernel<<<g3, 256, 0, stream>>>(attn, Wo, out);
}
